// HMLSTMOutput_6657199309448
// MI455X (gfx1250) — compile-verified, matrix operand order checked by reading, not run on hardware
//
#include <hip/hip_runtime.h>
#include <hip/hip_bf16.h>
#include <math.h>

// ---- problem dims (fixed by the reference) ----
#define NB    64
#define TT    512
#define NROWS (NB * TT)      // 32768
#define LL    3
#define INSZ  512
#define DD    (LL * INSZ)    // 1536
#define EE    1024
#define HH1   1024
#define HH2   1024
#define OO    512

typedef __bf16 v16bf __attribute__((ext_vector_type(16)));
typedef float  v8f   __attribute__((ext_vector_type(8)));
typedef int    v4i   __attribute__((ext_vector_type(4)));

union FragBF {
    v16bf v;
    uint4 q[2];
};

// CDNA5 async global->LDS copy path (ASYNCcnt), guarded so we fall back to
// the VGPR round-trip if the toolchain doesn't expose the builtin.
#if defined(__HIP_DEVICE_COMPILE__) && __has_builtin(__builtin_amdgcn_global_load_async_to_lds_b128)
#define USE_ASYNC 1
#else
#define USE_ASYNC 0
#endif

__device__ __forceinline__ void cp16(const __bf16* g, __bf16* s) {
#if USE_ASYNC
    __builtin_amdgcn_global_load_async_to_lds_b128(
        (v4i*)const_cast<__bf16*>(g), (v4i*)s, 0, 0);
#else
    *(uint4*)s = *(const uint4*)g;
#endif
}

__device__ __forceinline__ void async_wait_all() {
#if USE_ASYNC
#if __has_builtin(__builtin_amdgcn_s_wait_asynccnt)
    __builtin_amdgcn_s_wait_asynccnt(0);
#else
    asm volatile("s_wait_asynccnt 0x0" ::: "memory");
#endif
#endif
}

// ------------------------------------------------------------------
// prep kernels: f32 -> bf16 weight conversion / layout rearrangement
// ------------------------------------------------------------------
__global__ __launch_bounds__(256) void k_f32_to_bf16(const float* __restrict__ s,
                                                     __bf16* __restrict__ d, int n) {
    int i = blockIdx.x * 256 + threadIdx.x;
    if (i < n) d[i] = (__bf16)s[i];
}

// Wcat[e, l*INSZ + i] = We[l, e, i]   (E x D bf16, row-major)
__global__ __launch_bounds__(256) void k_build_wcat(const float* __restrict__ We,
                                                    __bf16* __restrict__ Wcat) {
    int idx = blockIdx.x * 256 + threadIdx.x;      // 0 .. E*D-1
    int e  = idx / DD;
    int dd = idx - e * DD;
    int l  = dd >> 9;          // / INSZ
    int i  = dd & (INSZ - 1);
    Wcat[idx] = (__bf16)We[((size_t)(l * EE + e)) * INSZ + i];
}

// beSum[e] = sum_l be[l, e]
__global__ __launch_bounds__(256) void k_besum(const float* __restrict__ be,
                                               float* __restrict__ bs) {
    int e = blockIdx.x * 256 + threadIdx.x;
    if (e < EE) bs[e] = be[e] + be[EE + e] + be[2 * EE + e];
}

// ------------------------------------------------------------------
// gate kernel: one wave32 per row.
//   g_l = sigmoid(sum_d x[n,d]*Wg[l,d]);  Xg[n,d] = bf16(x[n,d]*g[d/512])
// ------------------------------------------------------------------
__global__ __launch_bounds__(256) void k_gate(const float* __restrict__ x,
                                              const float* __restrict__ Wg,
                                              __bf16* __restrict__ Xg) {
    int row  = blockIdx.x * 8 + (threadIdx.x >> 5);
    int lane = threadIdx.x & 31;
    const float* h = x + (size_t)row * DD;

    float p0 = 0.f, p1 = 0.f, p2 = 0.f;
    for (int d = lane; d < DD; d += 32) {
        float hv = h[d];
        p0 += hv * Wg[d];
        p1 += hv * Wg[DD + d];
        p2 += hv * Wg[2 * DD + d];
    }
    for (int off = 16; off > 0; off >>= 1) {
        p0 += __shfl_xor(p0, off, 32);
        p1 += __shfl_xor(p1, off, 32);
        p2 += __shfl_xor(p2, off, 32);
    }
    float g0 = 1.f / (1.f + __expf(-p0));
    float g1 = 1.f / (1.f + __expf(-p1));
    float g2 = 1.f / (1.f + __expf(-p2));

    __bf16* o = Xg + (size_t)row * DD;
    for (int d = lane; d < DD; d += 32) {
        float g = (d < INSZ) ? g0 : ((d < 2 * INSZ) ? g1 : g2);
        o[d] = (__bf16)(h[d] * g);
    }
}

// ------------------------------------------------------------------
// tiled bf16 WMMA GEMM:  C[N x E] = act( A[N x K] * W[E x K]^T + bias )
//   block = 256 threads (8 waves as 4Mx2N), tile 128(M) x 128(N-out)
//   K-step 32, double-buffered LDS with async global->LDS copies
//   act: 0 = relu -> bf16 out, 1 = tanh -> bf16 out, 2 = none -> f32 out
// ------------------------------------------------------------------
#define LDSTRIDE 40   // 32 + 8 bf16 pad -> conflict-free b128 fragment reads

__global__ __launch_bounds__(256) void k_wmma_gemm(const __bf16* __restrict__ A,
                                                   const __bf16* __restrict__ W,
                                                   const float* __restrict__ bias,
                                                   __bf16* __restrict__ outBf,
                                                   float* __restrict__ outF,
                                                   int K, int E, int act) {
    __shared__ __bf16 As[2][128 * LDSTRIDE];
    __shared__ __bf16 Bs[2][128 * LDSTRIDE];

    const int tid     = threadIdx.x;
    const int lane    = tid & 31;
    const int wv      = tid >> 5;          // 0..7
    const int wm      = wv & 3;            // row group: rows wm*32 .. +31
    const int wn      = wv >> 2;           // col group: cols wn*64 .. +63
    const int lm      = lane & 15;
    const int khalf   = (lane < 16) ? 0 : 8;
    const int rowBase = blockIdx.y * 128;
    const int colBase = blockIdx.x * 128;

    // cooperative-copy coordinates: each thread owns 16 cols of one row
    const int cr = tid >> 1;               // 0..127
    const int cc = (tid & 1) * 16;         // col 0 or 16

    const __bf16* gA = A + (size_t)(rowBase + cr) * K + cc;
    const __bf16* gW = W + (size_t)(colBase + cr) * K + cc;
    __bf16* sA0 = &As[0][cr * LDSTRIDE + cc];
    __bf16* sB0 = &Bs[0][cr * LDSTRIDE + cc];
    __bf16* sA1 = &As[1][cr * LDSTRIDE + cc];
    __bf16* sB1 = &Bs[1][cr * LDSTRIDE + cc];

    v8f acc[2][4] = {};

    const int steps = K >> 5;

    // preload K-tile 0 into buffer 0
    cp16(gA,     sA0);  cp16(gA + 8, sA0 + 8);
    cp16(gW,     sB0);  cp16(gW + 8, sB0 + 8);
    async_wait_all();
    __syncthreads();

    for (int k = 0; k < steps; ++k) {
        const int cur = k & 1;

        // prefetch K-tile k+1 into the other buffer (overlaps with WMMA below)
        if (k + 1 < steps) {
            const __bf16* pa = gA + (size_t)(k + 1) * 32;
            const __bf16* pw = gW + (size_t)(k + 1) * 32;
            __bf16* da = cur ? sA0 : sA1;
            __bf16* db = cur ? sB0 : sB1;
            cp16(pa,     da);  cp16(pa + 8, da + 8);
            cp16(pw,     db);  cp16(pw + 8, db + 8);
        }

        // A fragments: rows wm*32 + {0..15, 16..31}; per-lane row M = lm,
        // K elements in two contiguous runs of 8 per the ISA layout.
        FragBF a0, a1;
        const __bf16* ab0 = &As[cur][(wm * 32 +      lm) * LDSTRIDE + khalf];
        const __bf16* ab1 = &As[cur][(wm * 32 + 16 + lm) * LDSTRIDE + khalf];
        a0.q[0] = *(const uint4*)(ab0);
        a0.q[1] = *(const uint4*)(ab0 + 16);
        a1.q[0] = *(const uint4*)(ab1);
        a1.q[1] = *(const uint4*)(ab1 + 16);

        #pragma unroll
        for (int j = 0; j < 4; ++j) {
            FragBF b;
            const __bf16* bb = &Bs[cur][(wn * 64 + j * 16 + lm) * LDSTRIDE + khalf];
            b.q[0] = *(const uint4*)(bb);
            b.q[1] = *(const uint4*)(bb + 16);
            acc[0][j] = __builtin_amdgcn_wmma_f32_16x16x32_bf16(
                false, a0.v, false, b.v, (short)0, acc[0][j], false, false);
            acc[1][j] = __builtin_amdgcn_wmma_f32_16x16x32_bf16(
                false, a1.v, false, b.v, (short)0, acc[1][j], false, false);
        }

        async_wait_all();
        __syncthreads();
    }

    // epilogue: C/D layout — lanes 0-15: M=v, N=lane; lanes 16-31: M=v+8, N=lane-16
    const int mofs = (lane >= 16) ? 8 : 0;
    #pragma unroll
    for (int mi = 0; mi < 2; ++mi) {
        #pragma unroll
        for (int j = 0; j < 4; ++j) {
            const int col = colBase + wn * 64 + j * 16 + lm;
            const float bv = bias[col];
            #pragma unroll
            for (int v = 0; v < 8; ++v) {
                const int row = rowBase + wm * 32 + mi * 16 + v + mofs;
                float val = acc[mi][j][v] + bv;
                if (act == 0)      val = fmaxf(val, 0.f);
                else if (act == 1) val = tanhf(val);
                if (act == 2) outF[(size_t)row * E + col] = val;
                else          outBf[(size_t)row * E + col] = (__bf16)val;
            }
        }
    }
}

// ------------------------------------------------------------------
// launch
// ------------------------------------------------------------------
extern "C" void kernel_launch(void* const* d_in, const int* in_sizes, int n_in,
                              void* d_out, int out_size, void* d_ws, size_t ws_size,
                              hipStream_t stream) {
    const float* x  = (const float*)d_in[0];
    const float* Wg = (const float*)d_in[1];
    const float* We = (const float*)d_in[2];
    const float* be = (const float*)d_in[3];
    const float* W1 = (const float*)d_in[4];
    const float* b1 = (const float*)d_in[5];
    const float* W2 = (const float*)d_in[6];
    const float* b2 = (const float*)d_in[7];
    const float* Wo = (const float*)d_in[8];
    const float* bo = (const float*)d_in[9];
    float* out = (float*)d_out;

    // workspace carve-up (256B aligned)
    char* ws = (char*)d_ws;
    auto carve = [&](size_t bytes) {
        char* p = ws;
        ws += (bytes + 255) & ~(size_t)255;
        return p;
    };
    __bf16* Wcat  = (__bf16*)carve((size_t)EE * DD * 2);    //  3 MiB
    __bf16* W1b   = (__bf16*)carve((size_t)HH1 * EE * 2);   //  2 MiB
    __bf16* W2b   = (__bf16*)carve((size_t)HH2 * HH1 * 2);  //  2 MiB
    __bf16* Wob   = (__bf16*)carve((size_t)OO * HH2 * 2);   //  1 MiB
    float*  beSum = (float*)carve((size_t)EE * 4);
    __bf16* Xg    = (__bf16*)carve((size_t)NROWS * DD * 2); // 96 MiB (reused for o1)
    __bf16* bufA  = (__bf16*)carve((size_t)NROWS * EE * 2); // 64 MiB (he, reused for o2)

    __bf16* he = bufA;
    __bf16* o1 = Xg;    // Xg dead after GEMM1
    __bf16* o2 = bufA;  // he dead after GEMM2

    // --- weight prep ---
    k_build_wcat<<<(EE * DD) / 256, 256, 0, stream>>>(We, Wcat);
    k_f32_to_bf16<<<(HH1 * EE) / 256, 256, 0, stream>>>(W1, W1b, HH1 * EE);
    k_f32_to_bf16<<<(HH2 * HH1) / 256, 256, 0, stream>>>(W2, W2b, HH2 * HH1);
    k_f32_to_bf16<<<(OO * HH2) / 256, 256, 0, stream>>>(Wo, Wob, OO * HH2);
    k_besum<<<EE / 256, 256, 0, stream>>>(be, beSum);

    // --- gates + gated bf16 activations ---
    k_gate<<<NROWS / 8, 256, 0, stream>>>(x, Wg, Xg);

    // --- GEMM1: he = relu(Xg @ Wcat^T + beSum)   [N x 1024] ---
    k_wmma_gemm<<<dim3(EE / 128, NROWS / 128), 256, 0, stream>>>(
        Xg, Wcat, beSum, he, nullptr, DD, EE, 0);

    // --- GEMM2: o1 = tanh(he @ W1^T + b1)        [N x 1024] ---
    k_wmma_gemm<<<dim3(HH1 / 128, NROWS / 128), 256, 0, stream>>>(
        he, W1b, b1, o1, nullptr, EE, HH1, 1);

    // --- GEMM3: o2 = tanh(o1 @ W2^T + b2)        [N x 1024] ---
    k_wmma_gemm<<<dim3(HH2 / 128, NROWS / 128), 256, 0, stream>>>(
        o1, W2b, b2, o2, nullptr, HH1, HH2, 1);

    // --- GEMM4: out = o2 @ Wo^T + bo             [N x 512, f32] ---
    k_wmma_gemm<<<dim3(OO / 128, NROWS / 128), 256, 0, stream>>>(
        o2, Wob, bo, nullptr, out, HH2, OO, 2);
}
